// T5Attention_51926154608988
// MI455X (gfx1250) — matrix operand order repaired and checked by reading, not compiled, not run on hardware
//
#include <hip/hip_runtime.h>
#include <hip/hip_bf16.h>
#include <stdint.h>

#define T_LEN   2048
#define D_MODEL 1024
#define NH      16
#define DK      64
#define HD      (NH*DK)          // 1024
#define RELSPAN (2*T_LEN - 1)    // 4095

typedef __attribute__((ext_vector_type(16))) __bf16 v16bf;
typedef __attribute__((ext_vector_type(8)))  float  v8f;

union Frag32 {              // 32-byte (16 x bf16) WMMA operand fragment
    v16bf v;
    uint4 q[2];
};

__device__ __forceinline__ v8f wmma_bf16(const v16bf& a, const v16bf& b, const v8f& c) {
    // D = A(16x32 bf16) x B(32x16 bf16) + C(16x16 f32)
    return __builtin_amdgcn_wmma_f32_16x16x32_bf16(false, a, false, b, (short)0, c, false, false);
}

__device__ __forceinline__ __bf16 f2bf(float f) { return (__bf16)f; }

// async DMA: 16 bytes global -> LDS, tracked by ASYNCcnt (no VGPR round-trip).
// LDS operand is the wave-relative LDS byte address = low 32 bits of the
// generic pointer (ISA 10.2: LDS aperture addresses truncate to addr[31:0]).
__device__ __forceinline__ void async_load_b128(const void* gptr, void* lptr) {
    unsigned lds = (unsigned)(uintptr_t)lptr;
    asm volatile("global_load_async_to_lds_b128 %0, %1, off"
                 :: "v"(lds), "v"((uint64_t)(uintptr_t)gptr)
                 : "memory");
}

// ---------------------------------------------------------------- prep kernels

__global__ void cast_f32_bf16_kernel(const float* __restrict__ src,
                                     __bf16* __restrict__ dst, int n) {
    int i = blockIdx.x * blockDim.x + threadIdx.x;
    if (i < n) dst[i] = f2bf(src[i]);
}

// src [K][N] f32  ->  dst [N][K] bf16  (B^T layout for WMMA B fragments)
__global__ void transpose_cast_kernel(const float* __restrict__ src,
                                      __bf16* __restrict__ dst, int K, int N) {
    int n = blockIdx.x * 16 + threadIdx.x;
    int k = blockIdx.y * 16 + threadIdx.y;
    if (n < N && k < K) dst[(size_t)n * K + k] = f2bf(src[(size_t)k * N + n]);
}

// packed-varlen metadata: seq_id[t], pos[t]  (seq_id = -1 for invalid tail)
__global__ void meta_kernel(const int* __restrict__ q_lens, int nseq,
                            int* __restrict__ seq_id, int* __restrict__ pos) {
    if (threadIdx.x != 0 || blockIdx.x != 0) return;
    int total = 0;
    for (int i = 0; i < nseq; i++) total += q_lens[i];
    if (total > T_LEN) total = T_LEN;
    int si = 0, st = 0, en = (nseq > 0) ? q_lens[0] : 0;
    for (int t = 0; t < T_LEN; t++) {
        if (t >= total) { seq_id[t] = -1; pos[t] = 0; continue; }
        while (si < nseq - 1 && t >= en) { st = en; si++; en += q_lens[si]; }
        seq_id[t] = si;
        pos[t] = t - st;
    }
}

// biasTab[h][rel + (T-1)] = rel_bias_table[bucket(rel)][h], rel = pos_k - pos_q
__global__ void bias_table_kernel(const float* __restrict__ tbl,
                                  float* __restrict__ biasTab) {
    int i = blockIdx.x * blockDim.x + threadIdx.x;
    if (i >= NH * RELSPAN) return;
    int h   = i / RELSPAN;
    int rel = (i % RELSPAN) - (T_LEN - 1);
    // bidirectional T5 bucket: 16 per direction, max_exact=8, max_distance=128
    int n = -rel;
    int ret = 0;
    if (n < 0) { ret = 16; n = -n; }
    int b;
    if (n < 8) {
        b = n;
    } else {
        float lv = __logf((float)n * 0.125f) * (8.0f / __logf(16.0f));
        int vl = 8 + (int)lv;
        b = vl < 15 ? vl : 15;
    }
    b += ret;
    biasTab[i] = tbl[b * NH + h];
}

// ---------------------------------------------------------------- QKV projection GEMM
// One wave per 32x64 macro-tile (2x4 register blocking): 8 WMMAs / 12 loads per k-step.
// mode 0/1: write [H][T][DK] bf16 (Q,K).  mode 2: write V transposed [H][DK][T] bf16.
__global__ void __launch_bounds__(128) gemm_proj_kernel(
    const __bf16* __restrict__ A,    // [M][K] bf16
    const __bf16* __restrict__ Bt,   // [N][K] bf16 (W^T)
    __bf16* __restrict__ Out, int mode) {
    const int M = T_LEN, N = HD, K = D_MODEL;
    int wid  = threadIdx.x >> 5;
    int lane = threadIdx.x & 31;
    int half = lane >> 4;
    int ln   = lane & 15;
    int tile = blockIdx.x * 4 + wid;
    const int ntn = N / 64;              // 16 macro-columns
    int mt = tile / ntn;
    int nt = tile % ntn;
    if (mt >= M / 32) return;
    int m0 = mt * 32, n0 = nt * 64;

    const __bf16* arow[2] = { A + (size_t)(m0 + ln) * K,
                              A + (size_t)(m0 + 16 + ln) * K };
    const __bf16* brow[4];
    #pragma unroll
    for (int j = 0; j < 4; j++) brow[j] = Bt + (size_t)(n0 + j * 16 + ln) * K;

    v8f acc[2][4] = {};
    #pragma unroll 2
    for (int k0 = 0; k0 < K; k0 += 32) {
        Frag32 af[2], bfg[4];
        #pragma unroll
        for (int i = 0; i < 2; i++) {
            const __bf16* ap = arow[i] + k0;
            af[i].q[0] = *(const uint4*)(ap + half * 16);
            af[i].q[1] = *(const uint4*)(ap + half * 16 + 8);
        }
        #pragma unroll
        for (int j = 0; j < 4; j++) {
            const __bf16* bp = brow[j] + k0 + half * 16;
            bfg[j].q[0] = *(const uint4*)(bp);
            bfg[j].q[1] = *(const uint4*)(bp + 8);
        }
        #pragma unroll
        for (int i = 0; i < 2; i++)
            #pragma unroll
            for (int j = 0; j < 4; j++)
                acc[i][j] = wmma_bf16(af[i].v, bfg[j].v, acc[i][j]);
    }

    #pragma unroll
    for (int j = 0; j < 4; j++) {
        int n = n0 + j * 16 + ln;
        int h = n >> 6, nd = n & 63;
        #pragma unroll
        for (int i = 0; i < 2; i++) {
            #pragma unroll
            for (int e = 0; e < 8; e++) {
                int m = m0 + i * 16 + e + half * 8;
                __bf16 val = f2bf(acc[i][j][e]);
                if (mode == 2) Out[((size_t)h * DK + nd) * T_LEN + m] = val;   // V^T
                else           Out[((size_t)h * T_LEN + m) * DK + nd] = val;   // Q,K
            }
        }
    }
}

// ---------------------------------------------------------------- flash attention
// Block = 4 waves on the same head, adjacent q-tiles. K/V blocks are staged into
// LDS via async global->LDS DMA with DOUBLE BUFFERING: each iteration issues the
// next block's DMAs, then uses a counted s_wait_asynccnt 0x4 so only the current
// buffer's (oldest, in-order) DMAs are waited on while the new ones fly.
__global__ void __launch_bounds__(128) attn_kernel(
    const __bf16* __restrict__ Qb,      // [H][T][DK]
    const __bf16* __restrict__ Kb,      // [H][T][DK]
    const __bf16* __restrict__ Vt,      // [H][DK][T]
    const float*  __restrict__ biasTab, // [H][RELSPAN]
    const int*    __restrict__ seq_id,
    const int*    __restrict__ pos,
    __bf16* __restrict__ AttnOut) {     // [T][HD]
    __shared__ __bf16 Ksh[2][32][64];   // K blocks   [buf][kv][d]   (8 KB)
    __shared__ __bf16 Vsh[2][64][32];   // V^T blocks [buf][d][kv]   (8 KB)
    __shared__ __bf16 Pb[4][16][32];    // per-wave P staging        (4 KB)

    int tid  = threadIdx.x;
    int wid  = tid >> 5;
    int lane = tid & 31;
    int half = lane >> 4;
    int ln   = lane & 15;
    int h    = blockIdx.x >> 5;         // 32 blocks per head
    int qblk = blockIdx.x & 31;
    int q0   = (qblk * 4 + wid) * 16;

    const __bf16* Kbase = Kb + (size_t)h * T_LEN * DK;
    const __bf16* Vbase = Vt + (size_t)h * DK * T_LEN;

    // issue the 4 per-thread DMAs (8 KB per block total) for one kv block
    auto stage = [&](int buf, int kv0) {
        #pragma unroll
        for (int c = 0; c < 2; c++) {
            int chunk = tid * 2 + c;                 // 0..255
            int kr = chunk >> 3, kc = (chunk & 7) * 8;
            async_load_b128(Kbase + (size_t)(kv0 + kr) * DK + kc, &Ksh[buf][kr][kc]);
            int vr = chunk >> 2, vc = (chunk & 3) * 8;
            async_load_b128(Vbase + (size_t)vr * T_LEN + kv0 + vc, &Vsh[buf][vr][vc]);
        }
    };

    // Q fragments for both K-steps (d_kv = 64 = 2 x 32)
    Frag32 qa[2];
    const __bf16* qrow = Qb + ((size_t)h * T_LEN + q0 + ln) * DK;
    #pragma unroll
    for (int s = 0; s < 2; s++) {
        qa[s].q[0] = *(const uint4*)(qrow + s * 32 + half * 16);
        qa[s].q[1] = *(const uint4*)(qrow + s * 32 + half * 16 + 8);
    }

    int sq[8], pq[8];
    #pragma unroll
    for (int e = 0; e < 8; e++) {
        int qi = q0 + e + half * 8;
        sq[e] = seq_id[qi];
        pq[e] = pos[qi];
    }
    const float* btab = biasTab + (size_t)h * RELSPAN + (T_LEN - 1);

    float mrow[8], lrow[8];
    v8f acc[4] = {};
    #pragma unroll
    for (int e = 0; e < 8; e++) { mrow[e] = -1e30f; lrow[e] = 0.0f; }

    stage(0, 0);                        // prologue: first block in flight
    int cur = 0;

    for (int kv0 = 0; kv0 < T_LEN; kv0 += 32) {
        // issue next block into the other buffer (its readers finished at the
        // barrier that ended the previous iteration); last iter wraps harmlessly
        int nkv = (kv0 + 32 < T_LEN) ? (kv0 + 32) : 0;
        stage(cur ^ 1, nkv);

        // wait until only the 4 just-issued DMAs remain outstanding:
        // async loads complete in order, so the current buffer has landed
        asm volatile("s_wait_asynccnt 0x4" ::: "memory");
        __syncthreads();                // all threads' current-buffer DMAs landed

        // ---- scores: two 16x16 tiles, each = 2 WMMAs over d_kv (K from LDS)
        v8f sc[2];
        #pragma unroll
        for (int j = 0; j < 2; j++) {
            v8f c = {};
            #pragma unroll
            for (int s = 0; s < 2; s++) {
                Frag32 kf;
                const __bf16* kp = &Ksh[cur][j * 16 + ln][s * 32 + half * 16];
                kf.q[0] = *(const uint4*)(kp);
                kf.q[1] = *(const uint4*)(kp + 8);
                c = wmma_bf16(qa[s].v, kf.v, c);
            }
            sc[j] = c;
        }

        // ---- bias + block-diagonal mask (fp32)
        float p[2][8];
        #pragma unroll
        for (int j = 0; j < 2; j++) {
            int kc2 = kv0 + j * 16 + ln;
            int sk = seq_id[kc2];
            int pk = pos[kc2];
            #pragma unroll
            for (int e = 0; e < 8; e++) {
                float s = sc[j][e];
                if (sq[e] >= 0 && sq[e] == sk) s += btab[pk - pq[e]];
                else                           s = -1e30f;
                p[j][e] = s;
            }
        }

        // ---- online softmax: row reductions across the 16-lane half-wave
        float mblk[8];
        #pragma unroll
        for (int e = 0; e < 8; e++) mblk[e] = fmaxf(p[0][e], p[1][e]);
        #pragma unroll
        for (int off = 1; off < 16; off <<= 1) {
            #pragma unroll
            for (int e = 0; e < 8; e++)
                mblk[e] = fmaxf(mblk[e], __shfl_xor(mblk[e], off, 32));
        }
        float lblk[8];
        #pragma unroll
        for (int e = 0; e < 8; e++) {
            float mn    = fmaxf(mrow[e], mblk[e]);
            float scale = __expf(mrow[e] - mn);
            mrow[e] = mn;
            lrow[e] *= scale;
            #pragma unroll
            for (int t = 0; t < 4; t++) acc[t][e] *= scale;
            p[0][e] = __expf(p[0][e] - mn);
            p[1][e] = __expf(p[1][e] - mn);
            lblk[e] = p[0][e] + p[1][e];
        }
        #pragma unroll
        for (int off = 1; off < 16; off <<= 1) {
            #pragma unroll
            for (int e = 0; e < 8; e++)
                lblk[e] += __shfl_xor(lblk[e], off, 32);
        }
        #pragma unroll
        for (int e = 0; e < 8; e++) lrow[e] += lblk[e];

        // ---- stage P (16x32 bf16) through LDS: C-layout writes, A-fragment reads
        #pragma unroll
        for (int j = 0; j < 2; j++)
            #pragma unroll
            for (int e = 0; e < 8; e++)
                Pb[wid][e + half * 8][j * 16 + ln] = f2bf(p[j][e]);

        // same-wave DS RAW: DS ops are in-order, make completion explicit
        asm volatile("s_wait_dscnt 0x0" ::: "memory");

        Frag32 pf;
        const __bf16* prow = &Pb[wid][ln][0];
        pf.q[0] = *(const uint4*)(prow + half * 16);
        pf.q[1] = *(const uint4*)(prow + half * 16 + 8);

        // ---- PV: 4 output d-tiles, B from V^T block in LDS
        #pragma unroll
        for (int t = 0; t < 4; t++) {
            Frag32 vf;
            const __bf16* vrow = &Vsh[cur][t * 16 + ln][half * 16];
            vf.q[0] = *(const uint4*)(vrow);
            vf.q[1] = *(const uint4*)(vrow + 8);
            acc[t] = wmma_bf16(pf.v, vf.v, acc[t]);
        }

        __syncthreads();                // all reads of buffer `cur` done ->
        cur ^= 1;                       // safe to overwrite it next iteration
    }

    // ---- epilogue: normalize, write attn output [T][H*DK] bf16
    #pragma unroll
    for (int t = 0; t < 4; t++) {
        #pragma unroll
        for (int e = 0; e < 8; e++) {
            int m = q0 + e + half * 8;
            float l = lrow[e];
            float v = acc[t][e] / (l > 0.0f ? l : 1.0f);
            AttnOut[(size_t)m * HD + h * DK + t * 16 + ln] = f2bf(v);
        }
    }
}

// ---------------------------------------------------------------- output projection GEMM
// Out[T][D] f32 = Attn[T][HD] bf16 x Wo ; Bt = Wo^T [D][HD] bf16; 2x4 blocking.
__global__ void __launch_bounds__(128) gemm_out_kernel(
    const __bf16* __restrict__ A,
    const __bf16* __restrict__ Bt,
    float* __restrict__ Out) {
    const int M = T_LEN, N = D_MODEL, K = HD;
    int wid  = threadIdx.x >> 5;
    int lane = threadIdx.x & 31;
    int half = lane >> 4;
    int ln   = lane & 15;
    int tile = blockIdx.x * 4 + wid;
    const int ntn = N / 64;
    int mt = tile / ntn;
    int nt = tile % ntn;
    if (mt >= M / 32) return;
    int m0 = mt * 32, n0 = nt * 64;

    const __bf16* arow[2] = { A + (size_t)(m0 + ln) * K,
                              A + (size_t)(m0 + 16 + ln) * K };
    const __bf16* brow[4];
    #pragma unroll
    for (int j = 0; j < 4; j++) brow[j] = Bt + (size_t)(n0 + j * 16 + ln) * K;

    v8f acc[2][4] = {};
    #pragma unroll 2
    for (int k0 = 0; k0 < K; k0 += 32) {
        Frag32 af[2], bfg[4];
        #pragma unroll
        for (int i = 0; i < 2; i++) {
            const __bf16* ap = arow[i] + k0;
            af[i].q[0] = *(const uint4*)(ap + half * 16);
            af[i].q[1] = *(const uint4*)(ap + half * 16 + 8);
        }
        #pragma unroll
        for (int j = 0; j < 4; j++) {
            const __bf16* bp = brow[j] + k0 + half * 16;
            bfg[j].q[0] = *(const uint4*)(bp);
            bfg[j].q[1] = *(const uint4*)(bp + 8);
        }
        #pragma unroll
        for (int i = 0; i < 2; i++)
            #pragma unroll
            for (int j = 0; j < 4; j++)
                acc[i][j] = wmma_bf16(af[i].v, bfg[j].v, acc[i][j]);
    }

    #pragma unroll
    for (int j = 0; j < 4; j++) {
        int n = n0 + j * 16 + ln;
        #pragma unroll
        for (int i = 0; i < 2; i++) {
            #pragma unroll
            for (int e = 0; e < 8; e++) {
                int m = m0 + i * 16 + e + half * 8;
                Out[(size_t)m * N + n] = acc[i][j][e];
            }
        }
    }
}

// ---------------------------------------------------------------- launch

extern "C" void kernel_launch(void* const* d_in, const int* in_sizes, int n_in,
                              void* d_out, int out_size, void* d_ws, size_t ws_size,
                              hipStream_t stream) {
    const float* x     = (const float*)d_in[0];
    const float* Wq    = (const float*)d_in[1];
    const float* Wk    = (const float*)d_in[2];
    const float* Wv    = (const float*)d_in[3];
    const float* Wo    = (const float*)d_in[4];
    const float* rbt   = (const float*)d_in[5];
    const int*   qlens = (const int*)d_in[6];
    int nseq = in_sizes[6];
    float* out = (float*)d_out;

    // workspace carve-out (~29 MB)
    char* ws = (char*)d_ws;
    size_t off = 0;
    auto carve = [&](size_t bytes) -> void* {
        void* p = ws + off;
        off += (bytes + 255) & ~(size_t)255;
        return p;
    };
    __bf16* Xb   = (__bf16*)carve((size_t)T_LEN * D_MODEL * 2);
    __bf16* Wqt  = (__bf16*)carve((size_t)HD * D_MODEL * 2);
    __bf16* Wkt  = (__bf16*)carve((size_t)HD * D_MODEL * 2);
    __bf16* Wvt  = (__bf16*)carve((size_t)HD * D_MODEL * 2);
    __bf16* Wot  = (__bf16*)carve((size_t)D_MODEL * HD * 2);
    __bf16* Qb   = (__bf16*)carve((size_t)NH * T_LEN * DK * 2);
    __bf16* Kb   = (__bf16*)carve((size_t)NH * T_LEN * DK * 2);
    __bf16* Vtb  = (__bf16*)carve((size_t)NH * DK * T_LEN * 2);
    __bf16* Attn = (__bf16*)carve((size_t)T_LEN * HD * 2);
    float*  biasTab = (float*)carve((size_t)NH * RELSPAN * 4);
    int*    seq_id  = (int*)carve((size_t)T_LEN * 4);
    int*    pos     = (int*)carve((size_t)T_LEN * 4);
    (void)ws_size;

    // --- prep
    {
        int n = T_LEN * D_MODEL;
        cast_f32_bf16_kernel<<<(n + 255) / 256, 256, 0, stream>>>(x, Xb, n);
    }
    dim3 tb(16, 16), tg(D_MODEL / 16, D_MODEL / 16);
    transpose_cast_kernel<<<tg, tb, 0, stream>>>(Wq, Wqt, D_MODEL, HD);
    transpose_cast_kernel<<<tg, tb, 0, stream>>>(Wk, Wkt, D_MODEL, HD);
    transpose_cast_kernel<<<tg, tb, 0, stream>>>(Wv, Wvt, D_MODEL, HD);
    transpose_cast_kernel<<<tg, tb, 0, stream>>>(Wo, Wot, HD, D_MODEL);
    meta_kernel<<<1, 32, 0, stream>>>(qlens, nseq, seq_id, pos);
    {
        int n = NH * RELSPAN;
        bias_table_kernel<<<(n + 255) / 256, 256, 0, stream>>>(rbt, biasTab);
    }

    // --- QKV projections (WMMA, 32x64 macro-tiles)
    int proj_blocks = (T_LEN / 32) * (HD / 64) / 4;   // 256
    gemm_proj_kernel<<<proj_blocks, 128, 0, stream>>>(Xb, Wqt, Qb, 0);
    gemm_proj_kernel<<<proj_blocks, 128, 0, stream>>>(Xb, Wkt, Kb, 1);
    gemm_proj_kernel<<<proj_blocks, 128, 0, stream>>>(Xb, Wvt, Vtb, 2);

    // --- flash attention (WMMA + double-buffered async global->LDS staging)
    int attn_blocks = NH * (T_LEN / 16) / 4;          // 512
    attn_kernel<<<attn_blocks, 128, 0, stream>>>(Qb, Kb, Vtb, biasTab, seq_id, pos, Attn);

    // --- output projection (WMMA, f32 out)
    int out_blocks = (T_LEN / 32) * (D_MODEL / 64) / 4;
    gemm_out_kernel<<<out_blocks, 128, 0, stream>>>(Attn, Wot, out);
}
